// SDPAttention3D_2259152798589
// MI455X (gfx1250) — matrix operand order repaired and checked by reading, not compiled, not run on hardware
//
#include <hip/hip_runtime.h>
#include <hip/hip_bf16.h>

// ---------------------------------------------------------------------------
// SDP attention block for MI455X (gfx1250, wave32, WMMA f32_16x16x32_f16)
//   x:(1,16,16,16,512) -> L=4096 tokens, C=512
//   Pipeline: f32->f16 convert -> fused QKV GEMM -> rmsnorm+rope ->
//             GQA flash attention (online softmax) -> output GEMM
// ---------------------------------------------------------------------------

typedef __attribute__((ext_vector_type(16))) _Float16 v16h;
typedef __attribute__((ext_vector_type(8)))  _Float16 v8h;
typedef __attribute__((ext_vector_type(4)))  _Float16 v4h;
typedef __attribute__((ext_vector_type(8)))  float    v8f;

#define L_TOK   4096
#define CWIDTH  512
#define NHEADS  8
#define NKV     2
#define HD      64
#define QKV_N   768          // 512 q + 128 k + 128 v

__device__ __forceinline__ v8f wmma16(v16h a, v16h b, v8f c) {
  // D = A(16x32 f16) * B(32x16 f16) + C(16x16 f32)
  return __builtin_amdgcn_wmma_f32_16x16x32_f16(
      /*neg_a=*/false, a, /*neg_b=*/false, b,
      /*c_mod=*/(short)0, c, /*reuse_a=*/false, /*reuse_b=*/false);
}

// A-matrix (16x32, f16) lane layout: lane l holds row M=l%16.
//   elems 0..7  : K = (l<16 ? 0 : 8)  + e
//   elems 8..15 : K = (l<16 ? 16: 24) + (e-8)
__device__ __forceinline__ v16h load_A_f16(const _Float16* __restrict__ base, int ld) {
  const int lane = threadIdx.x & 31;
  const _Float16* p = base + (lane & 15) * ld + ((lane >> 4) << 3);
  v8h lo = *(const v8h*)(p);
  v8h hi = *(const v8h*)(p + 16);
  v16h a;
#pragma unroll
  for (int e = 0; e < 8; ++e) { a[e] = lo[e]; a[8 + e] = hi[e]; }
  return a;
}

// B-matrix (32x16, f16) lane layout: lane l holds column N=l%16.
//   elems 0..7 : K = (l<16 ? 0 : 8) + e ; elems 8..15 : K = (l<16 ? 16 : 24) + (e-8)
// base points at B[k=0][n=0] where B[k][n] is stored src[n*ld + k].
__device__ __forceinline__ v16h load_B_f16(const _Float16* __restrict__ base, int ld) {
  const int lane = threadIdx.x & 31;
  const _Float16* p = base + (lane & 15) * ld + ((lane >> 4) << 3);
  v8h lo = *(const v8h*)(p);
  v8h hi = *(const v8h*)(p + 16);
  v16h b;
#pragma unroll
  for (int e = 0; e < 8; ++e) { b[e] = lo[e]; b[8 + e] = hi[e]; }
  return b;
}

// ---------------------------------------------------------------------------
// Kernel 0: vectorized f32 -> f16 convert (4 elems/thread)
// ---------------------------------------------------------------------------
__global__ __launch_bounds__(256) void cvt_f16(
    const float* __restrict__ in, _Float16* __restrict__ out, int n4) {
  const int i = blockIdx.x * 256 + threadIdx.x;
  if (i < n4) {
    const float4 v = ((const float4*)in)[i];
    v4h o = { (_Float16)v.x, (_Float16)v.y, (_Float16)v.z, (_Float16)v.w };
    ((v4h*)out)[i] = o;
  }
}

// ---------------------------------------------------------------------------
// Kernel 1: C(L x N f32) = A(L x 512 f16) @ Bw^T (Bw: N x 512 f16 rows)
// One wave computes a 32x64 strip (2 A-frags share each B-frag).
// grid = (N/64, L/128), block = 128 (4 waves).
// ---------------------------------------------------------------------------
__global__ __launch_bounds__(128) void gemm_f16(
    const _Float16* __restrict__ A, const _Float16* __restrict__ Bw,
    float* __restrict__ C, int N) {
  const int wave  = threadIdx.x >> 5;
  const int lane  = threadIdx.x & 31;
  const int mBase = (blockIdx.y * 4 + wave) * 32;
  const int nBlk  = blockIdx.x * 64;

  v8f acc[2][4] = {};
  for (int kc = 0; kc < CWIDTH; kc += 32) {
    const v16h a0 = load_A_f16(A + mBase * CWIDTH + kc, CWIDTH);
    const v16h a1 = load_A_f16(A + (mBase + 16) * CWIDTH + kc, CWIDTH);
#pragma unroll
    for (int t = 0; t < 4; ++t) {
      const v16h b = load_B_f16(Bw + (nBlk + t * 16) * CWIDTH + kc, CWIDTH);
      acc[0][t] = wmma16(a0, b, acc[0][t]);
      acc[1][t] = wmma16(a1, b, acc[1][t]);
    }
  }
  // C layout: lane holds (M = r + (lane>=16 ? 8:0), N = lane%16) in vgpr r
  const int rowOff = (lane >> 4) << 3;
  const int col    = lane & 15;
#pragma unroll
  for (int i = 0; i < 2; ++i)
#pragma unroll
    for (int t = 0; t < 4; ++t)
#pragma unroll
      for (int r = 0; r < 8; ++r)
        C[(mBase + i * 16 + r + rowOff) * N + nBlk + t * 16 + col] = acc[i][t][r];
}

// ---------------------------------------------------------------------------
// Kernel 2: per-(token, head) RMSNorm + RoPE; emits f16 Q (x 1/sqrt(hd)),
// f16 K (row-major) and f16 V^T (64 x L per kv head). One wave per item,
// lane i owns d=i and d=i+32 (exactly the rotate-half pairing).
// items: u = 0..7 -> q head u ; 8..9 -> k head u-8 ; 10..11 -> v head u-10
// ---------------------------------------------------------------------------
__global__ __launch_bounds__(256) void norm_rope(
    const float* __restrict__ Y1, const float* __restrict__ qw,
    const float* __restrict__ kw, _Float16* __restrict__ Qb,
    _Float16* __restrict__ Kb, _Float16* __restrict__ Vt) {
  const int wid  = (int)((blockIdx.x * 256u + threadIdx.x) >> 5);
  const int lane = threadIdx.x & 31;
  const int l    = wid / 12;
  const int u    = wid % 12;
  if (l >= L_TOK) return;

  const float* src;
  if (u < 8)       src = Y1 + l * QKV_N + u * 64;
  else if (u < 10) src = Y1 + l * QKV_N + 512 + (u - 8) * 64;
  else             src = Y1 + l * QKV_N + 640 + (u - 10) * 64;

  float a = src[lane];
  float b = src[lane + 32];

  if (u < 10) {
    // rmsnorm over hd=64
    float ss = a * a + b * b;
#pragma unroll
    for (int m = 1; m < 32; m <<= 1) ss += __shfl_xor(ss, m, 32);
    const float r = rsqrtf(ss * (1.0f / 64.0f) + 1e-6f);
    const float* w = (u < 8) ? qw : kw;
    a = a * r * w[lane];
    b = b * r * w[lane + 32];
    // rope: inv_freq = 10000^(-2i/64), pair (d=i, d=i+32)
    const float inv_freq = __expf(-(float)(2 * lane) * (9.210340371976184f / 64.0f));
    const float ang = (float)l * inv_freq;
    float s, c;
    __sincosf(ang, &s, &c);
    const float o1 = a * c - b * s;
    const float o2 = b * c + a * s;
    a = o1; b = o2;
  }

  if (u < 8) {
    a *= 0.125f; b *= 0.125f;      // fold 1/sqrt(hd) into q
    _Float16* dst = Qb + (u * L_TOK + l) * HD;
    dst[lane] = (_Float16)a;
    dst[lane + 32] = (_Float16)b;
  } else if (u < 10) {
    _Float16* dst = Kb + ((u - 8) * L_TOK + l) * HD;
    dst[lane] = (_Float16)a;
    dst[lane + 32] = (_Float16)b;
  } else {
    _Float16* dst = Vt + (u - 10) * HD * L_TOK;   // [d][l]
    dst[lane * L_TOK + l]        = (_Float16)a;
    dst[(lane + 32) * L_TOK + l] = (_Float16)b;
  }
}

// ---------------------------------------------------------------------------
// Kernel 3: flash attention. One wave owns 16 query rows of one head and
// streams keys in blocks of 64 with online softmax:
//   - row-max reduced across the 16-lane group once per 64 keys
//   - softmax denominator kept as per-lane partial, reduced once at the end
//   - P (16x64 f32, C layout) goes through a 2KB per-wave LDS tile to become
//     two A-layout f16 fragments
// grid = (NHEADS, L/(16*4)), block = 128 (4 waves).
// ---------------------------------------------------------------------------
__global__ __launch_bounds__(128) void flash_attn(
    const _Float16* __restrict__ Qb, const _Float16* __restrict__ Kb,
    const _Float16* __restrict__ Vt, _Float16* __restrict__ AO) {
  __shared__ _Float16 Plds[4][16 * 64];
  const int wave  = threadIdx.x >> 5;
  const int lane  = threadIdx.x & 31;
  const int h     = blockIdx.x;
  const int kvh   = h >> 2;                 // 8 q heads -> 2 kv heads
  const int qBase = (blockIdx.y * 4 + wave) * 16;

  const _Float16* Qh = Qb + (h * L_TOK + qBase) * HD;
  const _Float16* Kh = Kb + (kvh * L_TOK) * HD;
  const _Float16* Vh = Vt + kvh * HD * L_TOK;

  const v16h qa0 = load_A_f16(Qh, HD);       // hd 0..31
  const v16h qa1 = load_A_f16(Qh + 32, HD);  // hd 32..63

  v8f o0 = {}, o1 = {}, o2 = {}, o3 = {};
  float mI[8], lI[8];                        // lI = per-lane partial sums
#pragma unroll
  for (int r = 0; r < 8; ++r) { mI[r] = -1e30f; lI[r] = 0.0f; }

  _Float16* myP = &Plds[wave][0];
  const int rowOff = (lane >> 4) << 3;
  const int col    = lane & 15;

  for (int kb = 0; kb < L_TOK; kb += 64) {
    // scores: four 16-key tiles
    v8f s0 = {}, s1 = {}, s2 = {}, s3 = {};
    s0 = wmma16(qa0, load_B_f16(Kh + (kb +  0) * HD,      HD), s0);
    s0 = wmma16(qa1, load_B_f16(Kh + (kb +  0) * HD + 32, HD), s0);
    s1 = wmma16(qa0, load_B_f16(Kh + (kb + 16) * HD,      HD), s1);
    s1 = wmma16(qa1, load_B_f16(Kh + (kb + 16) * HD + 32, HD), s1);
    s2 = wmma16(qa0, load_B_f16(Kh + (kb + 32) * HD,      HD), s2);
    s2 = wmma16(qa1, load_B_f16(Kh + (kb + 32) * HD + 32, HD), s2);
    s3 = wmma16(qa0, load_B_f16(Kh + (kb + 48) * HD,      HD), s3);
    s3 = wmma16(qa1, load_B_f16(Kh + (kb + 48) * HD + 32, HD), s3);

#pragma unroll
    for (int r = 0; r < 8; ++r) {
      float mx = fmaxf(fmaxf(s0[r], s1[r]), fmaxf(s2[r], s3[r]));
      mx = fmaxf(mx, __shfl_xor(mx, 1, 32));
      mx = fmaxf(mx, __shfl_xor(mx, 2, 32));
      mx = fmaxf(mx, __shfl_xor(mx, 4, 32));
      mx = fmaxf(mx, __shfl_xor(mx, 8, 32));
      const float mNew = fmaxf(mI[r], mx);
      const float sf = __expf(mI[r] - mNew);   // row-uniform across lane group
      mI[r] = mNew;
      const float p0 = __expf(s0[r] - mNew);
      const float p1 = __expf(s1[r] - mNew);
      const float p2 = __expf(s2[r] - mNew);
      const float p3 = __expf(s3[r] - mNew);
      lI[r] = lI[r] * sf + ((p0 + p1) + (p2 + p3));
      o0[r] *= sf; o1[r] *= sf; o2[r] *= sf; o3[r] *= sf;
      // C layout -> row-major [16][64] staging for A-layout reload
      const int row = (r + rowOff) * 64;
      myP[row + col]      = (_Float16)p0;
      myP[row + 16 + col] = (_Float16)p1;
      myP[row + 32 + col] = (_Float16)p2;
      myP[row + 48 + col] = (_Float16)p3;
    }
    asm volatile("s_wait_dscnt 0" ::: "memory");
    const v16h pa0 = load_A_f16(myP, 64);        // keys kb..kb+31
    const v16h pa1 = load_A_f16(myP + 32, 64);   // keys kb+32..kb+63
    // out += P(16x64) @ V(64x64): B from V^T rows (contiguous along keys)
    o0 = wmma16(pa0, load_B_f16(Vh +  0 * L_TOK + kb,      L_TOK), o0);
    o0 = wmma16(pa1, load_B_f16(Vh +  0 * L_TOK + kb + 32, L_TOK), o0);
    o1 = wmma16(pa0, load_B_f16(Vh + 16 * L_TOK + kb,      L_TOK), o1);
    o1 = wmma16(pa1, load_B_f16(Vh + 16 * L_TOK + kb + 32, L_TOK), o1);
    o2 = wmma16(pa0, load_B_f16(Vh + 32 * L_TOK + kb,      L_TOK), o2);
    o2 = wmma16(pa1, load_B_f16(Vh + 32 * L_TOK + kb + 32, L_TOK), o2);
    o3 = wmma16(pa0, load_B_f16(Vh + 48 * L_TOK + kb,      L_TOK), o3);
    o3 = wmma16(pa1, load_B_f16(Vh + 48 * L_TOK + kb + 32, L_TOK), o3);
  }

#pragma unroll
  for (int r = 0; r < 8; ++r) {
    // final reduction of the per-lane partial softmax denominator
    float ls = lI[r];
    ls += __shfl_xor(ls, 1, 32);
    ls += __shfl_xor(ls, 2, 32);
    ls += __shfl_xor(ls, 4, 32);
    ls += __shfl_xor(ls, 8, 32);
    const float inv = 1.0f / ls;
    _Float16* dst = AO + (qBase + r + rowOff) * CWIDTH + h * HD;
    dst[col]      = (_Float16)(o0[r] * inv);
    dst[16 + col] = (_Float16)(o1[r] * inv);
    dst[32 + col] = (_Float16)(o2[r] * inv);
    dst[48 + col] = (_Float16)(o3[r] * inv);
  }
}

// ---------------------------------------------------------------------------
extern "C" void kernel_launch(void* const* d_in, const int* in_sizes, int n_in,
                              void* d_out, int out_size, void* d_ws, size_t ws_size,
                              hipStream_t stream) {
  const float* x  = (const float*)d_in[0];
  const float* Wq = (const float*)d_in[1];
  const float* Wk = (const float*)d_in[2];
  const float* Wv = (const float*)d_in[3];
  const float* qw = (const float*)d_in[4];
  const float* kw = (const float*)d_in[5];
  const float* Wo = (const float*)d_in[6];
  float* out = (float*)d_out;

  char* ws = (char*)d_ws;
  _Float16* Xh  = (_Float16*)(ws);                     // L*512   f16 =  4 MB
  _Float16* Wh  = (_Float16*)(ws + 4194304);           // 768*512 f16 = .75 MB
  _Float16* Woh = (_Float16*)(ws + 4980736);           // 512*512 f16 = .5 MB
  float*    Y1  = (float*)   (ws + 5505024);           // L*768   f32 = 12 MB
  _Float16* Qb  = (_Float16*)(ws + 18087936);          // 8*L*64  f16 =  4 MB
  _Float16* Kb  = (_Float16*)(ws + 22282240);          // 2*L*64  f16 =  1 MB
  _Float16* Vt  = (_Float16*)(ws + 23330816);          // 2*64*L  f16 =  1 MB
  _Float16* AO  = (_Float16*)(ws + 24379392);          // L*512   f16 =  4 MB

  // 0) one-shot f32 -> f16 converts (removes cvt from all GEMM inner loops)
  cvt_f16<<<dim3(2048), 256, 0, stream>>>(x,  Xh,              2097152 / 4);
  cvt_f16<<<dim3(256),  256, 0, stream>>>(Wq, Wh,               262144 / 4);
  cvt_f16<<<dim3(64),   256, 0, stream>>>(Wk, Wh + 512 * 512,    65536 / 4);
  cvt_f16<<<dim3(64),   256, 0, stream>>>(Wv, Wh + 640 * 512,    65536 / 4);
  cvt_f16<<<dim3(256),  256, 0, stream>>>(Wo, Woh,              262144 / 4);

  // 1) fused QKV projection: Y1 = Xh @ Wh^T  (branch-free, pure f16 loads)
  gemm_f16<<<dim3(QKV_N / 64, L_TOK / 128), 128, 0, stream>>>(Xh, Wh, Y1, QKV_N);
  // 2) rmsnorm + rope + layout change (Q scaled, K row-major, V transposed)
  norm_rope<<<dim3((L_TOK * 12) / 8), 256, 0, stream>>>(Y1, qw, kw, Qb, Kb, Vt);
  // 3) GQA flash attention
  flash_attn<<<dim3(NHEADS, L_TOK / 64), 128, 0, stream>>>(Qb, Kb, Vt, AO);
  // 4) output projection
  gemm_f16<<<dim3(CWIDTH / 64, L_TOK / 128), 128, 0, stream>>>(AO, Woh, out, CWIDTH);
}
